// SwinTransformerBlock3D_15874199126215
// MI455X (gfx1250) — compile-verified, matrix operand order checked by reading, not run on hardware
//
#include <hip/hip_runtime.h>
#include <hip/hip_bf16.h>

typedef __attribute__((ext_vector_type(16))) __bf16 v16bf;
typedef __attribute__((ext_vector_type(8)))  float  v8f;

#define USHORT unsigned short

// ---------- helpers ----------
__device__ __forceinline__ USHORT f2bf(float x) {
    unsigned int u = __float_as_uint(x);
    unsigned int r = u + 0x7FFFu + ((u >> 16) & 1u);
    return (USHORT)(r >> 16);
}

union Frag32B { v16bf v; uint4 q[2]; };

// load one 16x(16|32) bf16 fragment: lane's 16 bf16 are contiguous (32B)
__device__ __forceinline__ v16bf ld_frag(const USHORT* fragBase, int lane) {
    Frag32B f;
    const uint4* p = (const uint4*)(fragBase + lane * 16);
    f.q[0] = p[0];
    f.q[1] = p[1];
    return f.v;
}

__device__ __forceinline__ v8f wmma_bf16(v16bf a, v16bf b, v8f c) {
    return __builtin_amdgcn_wmma_f32_16x16x32_bf16(false, a, false, b, (short)0, c,
                                                   false, false);
}

// k index inside a 32-wide K chunk for A/B fragment element e, lane half hi
// k = e + 8*(e>>3) + 8*hi  ;  inverse: hi=(k>>3)&1, e=(k&7)+8*(k>>4)
__device__ __forceinline__ int kOfE(int e, int hi) { return e + 8 * (e >> 3) + 8 * hi; }

__device__ __forceinline__ int tokBase(int bb, int dd, int hb, int wb, int m) {
    int wh = m / 7, wwp = m - wh * 7;
    int hh = hb * 7 + wh, wg = wb * 7 + wwp;
    return (((bb * 8 + dd) * 56 + hh) * 56 + wg) * 256;
}

// ---------- kernel 0: weight repack to B-fragment bf16 ----------
// layout per matrix: frag(nt,kc) at ((nt*KC+kc)*512) ushorts; element (lane,e):
//   value = W[nt*16 + (lane&15)][kc*32 + kOfE(e, lane>>4)]
__global__ void __launch_bounds__(256)
pack_kernel(const float* qkv_w, const float* proj_w, const float* fc1_w,
            const float* fc2_w, USHORT* qkvB, USHORT* projB, USHORT* fc1B,
            USHORT* fc2B) {
    int gid = blockIdx.x * 256 + threadIdx.x;
    const int n0 = 196608, n1 = n0 + 65536, n2 = n1 + 262144, n3 = n2 + 262144;
    if (gid >= n3) return;
    const float* W;
    USHORT* Dst;
    int IN, KC, e;
    if (gid < n0)      { W = qkv_w;  Dst = qkvB; IN = 256;  KC = 8;  e = gid; }
    else if (gid < n1) { W = proj_w; Dst = projB; IN = 256; KC = 8;  e = gid - n0; }
    else if (gid < n2) { W = fc1_w;  Dst = fc1B; IN = 256;  KC = 8;  e = gid - n1; }
    else               { W = fc2_w;  Dst = fc2B; IN = 1024; KC = 32; e = gid - n2; }
    int frag = e >> 9, w = e & 511, ln = w >> 4, el = w & 15;
    int nt = frag / KC, kc = frag - nt * KC;
    int row = nt * 16 + (ln & 15);
    int col = kc * 32 + kOfE(el, ln >> 4);
    Dst[e] = f2bf(W[row * IN + col]);
}

// ---------- kernel 1: CPB MLP -> 16*sigmoid bias table (8,49,49) ----------
__global__ void __launch_bounds__(256)
cpb_kernel(const float* tab3, const float* w1, const float* b1, const float* w2,
           const int* rpi, float* rpb) {
    __shared__ float tab[169 * 8];
    int tid = threadIdx.x;
    for (int it = tid; it < 169 * 8; it += 256) {
        int e = it >> 3, h = it & 7;
        float c0 = tab3[e * 3 + 0], c1 = tab3[e * 3 + 1], c2 = tab3[e * 3 + 2];
        float acc = 0.f;
        for (int i = 0; i < 512; ++i) {
            float hv = fmaxf(c0 * w1[i * 3] + c1 * w1[i * 3 + 1] + c2 * w1[i * 3 + 2] +
                                 b1[i],
                             0.f);
            acc += hv * w2[h * 512 + i];
        }
        tab[e * 8 + h] = acc;
    }
    __syncthreads();
    for (int it = tid; it < 8 * 49 * 49; it += 256) {
        int h = it / 2401, rem = it - h * 2401;
        float v = tab[rpi[rem] * 8 + h];
        rpb[it] = 16.f / (1.f + expf(-v));
    }
}

// ---------- kernel 2: windowed attention + proj + LN1 + residual ----------
// one workgroup (256 thr = 8 waves) per window (2048 windows)
#define XA_OFF 0          // 4mt x 8kc x 1KB           = 32768
#define QA_OFF 32768      // q A-frags / later out OA  = 32768
#define KB_OFF 65536      // k B-frags                 = 32768
#define VB_OFF 98304      // v B-frags                 = 32768
#define PA_OFF 131072     // probs A-frags 8h*4mt*2kc  = 65536
#define S_OFF  196608     // f32 staging (25600 f)     = 102400
#define MU_OFF 299008     // mu[64], rs[64]            = 512
#define SMEM2  299520

__global__ void __launch_bounds__(256)
attn_kernel(const float* __restrict__ x, const float* logit_scale,
            const float* proj_b, const float* norm1_g, const float* norm1_b,
            const float* q_bias, const float* v_bias, const float* __restrict__ rpb,
            const USHORT* __restrict__ qkvB, const USHORT* __restrict__ projB,
            float* __restrict__ out) {
    extern __shared__ char smem[];
    USHORT* XA = (USHORT*)(smem + XA_OFF);
    USHORT* QA = (USHORT*)(smem + QA_OFF);
    USHORT* KB = (USHORT*)(smem + KB_OFF);
    USHORT* VB = (USHORT*)(smem + VB_OFF);
    USHORT* PA = (USHORT*)(smem + PA_OFF);
    USHORT* OA = QA;  // aliased after attention scores are done
    float* S = (float*)(smem + S_OFF);
    float* Sq = S;
    float* Sk = S + 12544;
    float* SC = S;  // aliased after normalize
    float* P = S;   // aliased after softmax/PV
    float* MU = (float*)(smem + MU_OFF);
    float* RS = MU + 64;

    const int tid = threadIdx.x, lane = tid & 31, wv = tid >> 5;
    const int hi = lane >> 4, l15 = lane & 15;
    const int wi = blockIdx.x;
    const int wb = wi & 7, hb = (wi >> 3) & 7, dd = (wi >> 6) & 7, bb = wi >> 9;

    // --- phase A: stage x window as bf16 A-fragments (rows >=49 zero) ---
    for (int idx = tid; idx < 4 * 8 * 512; idx += 256) {
        int f = idx >> 9, w = idx & 511, ln = w >> 4, e = w & 15;
        int mt = f >> 3, kc = f & 7;
        int m = (ln & 15) + 16 * mt;
        int k = kc * 32 + kOfE(e, ln >> 4);
        float v = (m < 49) ? x[tokBase(bb, dd, hb, wb, m) + k] : 0.f;
        XA[idx] = f2bf(v);
    }
    __syncthreads();

    // --- phase B: qkv GEMM (wave wv handles n-tiles 6wv..6wv+5) ---
    for (int ntl = 0; ntl < 6; ++ntl) {
        int ntg = wv * 6 + ntl;
        int seg = ntg >> 4;  // 0:q 1:k 2:v
        for (int mt = 0; mt < 4; ++mt) {
            v8f acc = {0.f, 0.f, 0.f, 0.f, 0.f, 0.f, 0.f, 0.f};
            for (int kc = 0; kc < 8; ++kc) {
                v16bf a = ld_frag(XA + (mt * 8 + kc) * 512, lane);
                v16bf b = ld_frag(qkvB + (ntg * 8 + kc) * 512, lane);
                acc = wmma_bf16(a, b, acc);
            }
            int ncol = ntg * 16 + l15;
            float bias = (seg == 0) ? q_bias[ncol] : (seg == 2 ? v_bias[ncol - 512] : 0.f);
            for (int r = 0; r < 8; ++r) {
                float vv = acc[r] + bias;
                int m = r + 8 * hi + 16 * mt;
                if (seg == 0) {
                    if (m < 49) Sq[m * 256 + ncol] = vv;
                } else if (seg == 1) {
                    if (m < 49) Sk[m * 256 + (ncol - 256)] = vv;
                } else {
                    int c = ncol - 512, h = c >> 5, d = c & 31;
                    int kk = m & 31, h2 = (kk >> 3) & 1, e2 = (kk & 7) + 8 * (kk >> 4);
                    int ln2 = (d & 15) + 16 * h2;
                    VB[(((h * 2 + (d >> 4)) * 2) + (m >> 5)) * 512 + ln2 * 16 + e2] =
                        f2bf(vv);
                }
            }
        }
    }
    __syncthreads();

    // --- phase C: l2-normalize q,k rows -> QA / KB fragments ---
    for (int it = tid; it < 784; it += 256) {
        int kind = it >= 392;
        int r = kind ? it - 392 : it;
        int m = r >> 3, h = r & 7;
        const float* src = (kind ? Sk : Sq) + m * 256 + h * 32;
        float tmp[32], ss = 0.f;
#pragma unroll
        for (int d = 0; d < 32; ++d) { tmp[d] = src[d]; ss += tmp[d] * tmp[d]; }
        float inv = 1.f / fmaxf(sqrtf(ss), 1e-12f);
        USHORT* dst = kind ? KB : QA;
        int fo = (h * 4 + (m >> 4)) * 512;
#pragma unroll
        for (int d = 0; d < 32; ++d) {
            int h2 = (d >> 3) & 1, e2 = (d & 7) + 8 * (d >> 4);
            dst[fo + ((m & 15) + 16 * h2) * 16 + e2] = f2bf(tmp[d] * inv);
        }
    }
    __syncthreads();

    // --- phase D: scores = qn @ kn^T (wave = head) ---
    {
        int h = wv;
        for (int mt = 0; mt < 4; ++mt)
            for (int nt = 0; nt < 4; ++nt) {
                v8f acc = {0.f, 0.f, 0.f, 0.f, 0.f, 0.f, 0.f, 0.f};
                v16bf a = ld_frag(QA + (h * 4 + mt) * 512, lane);
                v16bf b = ld_frag(KB + (h * 4 + nt) * 512, lane);
                acc = wmma_bf16(a, b, acc);
                for (int r = 0; r < 8; ++r) {
                    int m = r + 8 * hi + 16 * mt;
                    if (m < 49) SC[h * 3136 + m * 64 + l15 + 16 * nt] = acc[r];
                }
            }
    }
    __syncthreads();

    // --- phase E: softmax rows (+logit scale, +CPB bias) -> PA bf16 frags ---
    {
        int h = wv;
        float sc = expf(fminf(logit_scale[h], 4.6051702f));
        for (int m = lane; m < 49; m += 32) {
            float* row = SC + h * 3136 + m * 64;
            const float* rr = rpb + h * 2401 + m * 49;
            float mx = -1e30f;
            for (int j = 0; j < 49; ++j) {
                float v = row[j] * sc + rr[j];
                row[j] = v;
                mx = fmaxf(mx, v);
            }
            float sum = 0.f;
            for (int j = 0; j < 49; ++j) {
                float e = expf(row[j] - mx);
                row[j] = e;
                sum += e;
            }
            float inv = 1.f / sum;
            for (int j = 0; j < 64; ++j) {
                float p = (j < 49) ? row[j] * inv : 0.f;
                int kk = j & 31, h2 = (kk >> 3) & 1, e2 = (kk & 7) + 8 * (kk >> 4);
                PA[((h * 4 + (m >> 4)) * 2 + (j >> 5)) * 512 + ((m & 15) + 16 * h2) * 16 +
                   e2] = f2bf(p);
            }
        }
    }
    __syncthreads();

    // --- phase F: out = P @ V (wave = head) -> OA A-frags (bf16) ---
    {
        int h = wv;
        for (int mt = 0; mt < 4; ++mt)
            for (int dt = 0; dt < 2; ++dt) {
                v8f acc = {0.f, 0.f, 0.f, 0.f, 0.f, 0.f, 0.f, 0.f};
                for (int kc = 0; kc < 2; ++kc) {
                    v16bf a = ld_frag(PA + ((h * 4 + mt) * 2 + kc) * 512, lane);
                    v16bf b = ld_frag(VB + ((h * 2 + dt) * 2 + kc) * 512, lane);
                    acc = wmma_bf16(a, b, acc);
                }
                int c = h * 32 + dt * 16 + l15;
                int kk = c & 31, h2 = (kk >> 3) & 1, e2 = (kk & 7) + 8 * (kk >> 4);
                for (int r = 0; r < 8; ++r) {
                    int m = r + 8 * hi + 16 * mt;
                    OA[((m >> 4) * 8 + (c >> 5)) * 512 + ((m & 15) + 16 * h2) * 16 + e2] =
                        f2bf(acc[r]);
                }
            }
    }
    __syncthreads();

    // --- phase G: proj GEMM (wave wv -> n-tiles 2wv, 2wv+1) ---
    for (int ntl = 0; ntl < 2; ++ntl) {
        int ntg = wv * 2 + ntl;
        for (int mt = 0; mt < 4; ++mt) {
            v8f acc = {0.f, 0.f, 0.f, 0.f, 0.f, 0.f, 0.f, 0.f};
            for (int kc = 0; kc < 8; ++kc) {
                v16bf a = ld_frag(OA + (mt * 8 + kc) * 512, lane);
                v16bf b = ld_frag(projB + (ntg * 8 + kc) * 512, lane);
                acc = wmma_bf16(a, b, acc);
            }
            int n = ntg * 16 + l15;
            float pb = proj_b[n];
            for (int r = 0; r < 8; ++r) {
                int m = r + 8 * hi + 16 * mt;
                if (m < 49) P[m * 256 + n] = acc[r] + pb;
            }
        }
    }
    __syncthreads();

    // --- phase H: y = x + LN(proj_out) -> d_out ---
    if (tid < 49) {
        int m = tid;
        float s = 0.f;
        for (int c = 0; c < 256; ++c) s += P[m * 256 + c];
        float mu = s * (1.f / 256.f), var = 0.f;
        for (int c = 0; c < 256; ++c) {
            float d = P[m * 256 + c] - mu;
            var += d * d;
        }
        MU[m] = mu;
        RS[m] = rsqrtf(var * (1.f / 256.f) + 1e-5f);
    }
    __syncthreads();
    for (int idx = tid; idx < 49 * 256; idx += 256) {
        int m = idx >> 8, c = idx & 255;
        float val = (P[m * 256 + c] - MU[m]) * RS[m] * norm1_g[c] + norm1_b[c];
        int ga = tokBase(bb, dd, hb, wb, m) + c;
        out[ga] = x[ga] + val;
    }
}

// ---------- kernel 3: MLP + LN2 + residual (in place on d_out) ----------
#define MXA_OFF 0          // 2mt x 8kc x 1KB  = 16384
#define MHA_OFF 16384      // 2mt x 32kc x 1KB = 65536
#define MYS_OFF 81920      // 32x256 f32       = 32768
#define MMU_OFF 114688     // mu/rs            = 512
#define SMEM3   115200

__global__ void __launch_bounds__(256)
mlp_kernel(const USHORT* __restrict__ fc1B, const USHORT* __restrict__ fc2B,
           const float* fc1_b, const float* fc2_b, const float* g2, const float* b2,
           float* __restrict__ out) {
    extern __shared__ char smem[];
    USHORT* XA = (USHORT*)(smem + MXA_OFF);
    USHORT* HA = (USHORT*)(smem + MHA_OFF);
    float* YS = (float*)(smem + MYS_OFF);
    float* MU = (float*)(smem + MMU_OFF);
    float* RS = MU + 32;

    const int tid = threadIdx.x, lane = tid & 31, wv = tid >> 5;
    const int hi = lane >> 4, l15 = lane & 15;
    const int tok0 = blockIdx.x * 32;

    // stage x1 (current d_out) -> A frags
    for (int idx = tid; idx < 2 * 8 * 512; idx += 256) {
        int f = idx >> 9, w = idx & 511, ln = w >> 4, e = w & 15;
        int mt = f >> 3, kc = f & 7;
        int m = (ln & 15) + 16 * mt;
        int k = kc * 32 + kOfE(e, ln >> 4);
        XA[idx] = f2bf(out[(tok0 + m) * 256 + k]);
    }
    __syncthreads();

    // fc1 + GELU -> HA frags (wave wv: n-tiles 8wv..8wv+7 of 64)
    for (int ntl = 0; ntl < 8; ++ntl) {
        int ntg = wv * 8 + ntl;
        for (int mt = 0; mt < 2; ++mt) {
            v8f acc = {0.f, 0.f, 0.f, 0.f, 0.f, 0.f, 0.f, 0.f};
            for (int kc = 0; kc < 8; ++kc) {
                v16bf a = ld_frag(XA + (mt * 8 + kc) * 512, lane);
                v16bf b = ld_frag(fc1B + (ntg * 8 + kc) * 512, lane);
                acc = wmma_bf16(a, b, acc);
            }
            int col = ntg * 16 + l15;
            float bias = fc1_b[col];
            int kk = col & 31, h2 = (kk >> 3) & 1, e2 = (kk & 7) + 8 * (kk >> 4);
            for (int r = 0; r < 8; ++r) {
                int m = r + 8 * hi + 16 * mt;
                float xv = acc[r] + bias;
                float ge = 0.5f * xv * (1.f + erff(xv * 0.70710678f));
                HA[(mt * 32 + (col >> 5)) * 512 + ((m & 15) + 16 * h2) * 16 + e2] =
                    f2bf(ge);
            }
        }
    }
    __syncthreads();

    // fc2 -> YS (wave wv: n-tiles 2wv, 2wv+1 of 16)
    for (int ntl = 0; ntl < 2; ++ntl) {
        int ntg = wv * 2 + ntl;
        for (int mt = 0; mt < 2; ++mt) {
            v8f acc = {0.f, 0.f, 0.f, 0.f, 0.f, 0.f, 0.f, 0.f};
            for (int kc = 0; kc < 32; ++kc) {
                v16bf a = ld_frag(HA + (mt * 32 + kc) * 512, lane);
                v16bf b = ld_frag(fc2B + (ntg * 32 + kc) * 512, lane);
                acc = wmma_bf16(a, b, acc);
            }
            int n = ntg * 16 + l15;
            float bias = fc2_b[n];
            for (int r = 0; r < 8; ++r) {
                int m = r + 8 * hi + 16 * mt;
                YS[m * 256 + n] = acc[r] + bias;
            }
        }
    }
    __syncthreads();

    // LN2 + residual
    if (tid < 32) {
        int m = tid;
        float s = 0.f;
        for (int c = 0; c < 256; ++c) s += YS[m * 256 + c];
        float mu = s * (1.f / 256.f), var = 0.f;
        for (int c = 0; c < 256; ++c) {
            float d = YS[m * 256 + c] - mu;
            var += d * d;
        }
        MU[m] = mu;
        RS[m] = rsqrtf(var * (1.f / 256.f) + 1e-5f);
    }
    __syncthreads();
    for (int idx = tid; idx < 32 * 256; idx += 256) {
        int m = idx >> 8, c = idx & 255;
        int ga = (tok0 + m) * 256 + c;
        float y = (YS[m * 256 + c] - MU[m]) * RS[m] * g2[c] + b2[c];
        out[ga] = out[ga] + y;
    }
}

// ---------- host launch ----------
extern "C" void kernel_launch(void* const* d_in, const int* in_sizes, int n_in,
                              void* d_out, int out_size, void* d_ws, size_t ws_size,
                              hipStream_t stream) {
    (void)in_sizes; (void)n_in; (void)out_size; (void)ws_size;
    const float* x          = (const float*)d_in[0];
    const float* qkv_w      = (const float*)d_in[1];
    const float* q_bias     = (const float*)d_in[2];
    const float* v_bias     = (const float*)d_in[3];
    const float* logit_sc   = (const float*)d_in[4];
    const float* cpb_w1     = (const float*)d_in[5];
    const float* cpb_b1     = (const float*)d_in[6];
    const float* cpb_w2     = (const float*)d_in[7];
    const float* proj_w     = (const float*)d_in[8];
    const float* proj_b     = (const float*)d_in[9];
    const float* norm1_g    = (const float*)d_in[10];
    const float* norm1_b    = (const float*)d_in[11];
    const float* fc1_w      = (const float*)d_in[12];
    const float* fc1_b      = (const float*)d_in[13];
    const float* fc2_w      = (const float*)d_in[14];
    const float* fc2_b      = (const float*)d_in[15];
    const float* norm2_g    = (const float*)d_in[16];
    const float* norm2_b    = (const float*)d_in[17];
    const float* rel_tab    = (const float*)d_in[18];
    const int*   rel_idx    = (const int*)d_in[19];

    char* ws = (char*)d_ws;
    float* rpb   = (float*)ws;                       // 76832 B
    USHORT* qkvB = (USHORT*)(ws + 77824);            // 393216 B
    USHORT* projB = (USHORT*)(ws + 471040);          // 131072 B
    USHORT* fc1B = (USHORT*)(ws + 602112);           // 524288 B
    USHORT* fc2B = (USHORT*)(ws + 1126400);          // 524288 B

    pack_kernel<<<3072, 256, 0, stream>>>(qkv_w, proj_w, fc1_w, fc2_w, qkvB, projB,
                                          fc1B, fc2B);
    cpb_kernel<<<1, 256, 0, stream>>>(rel_tab, cpb_w1, cpb_b1, cpb_w2, rel_idx, rpb);
    attn_kernel<<<2048, 256, SMEM2, stream>>>(x, logit_sc, proj_b, norm1_g, norm1_b,
                                              q_bias, v_bias, rpb, qkvB, projB,
                                              (float*)d_out);
    mlp_kernel<<<3136, 256, SMEM3, stream>>>(fc1B, fc2B, fc1_b, fc2_b, norm2_g,
                                             norm2_b, (float*)d_out);
}
